// Evaluate_67430986547241
// MI455X (gfx1250) — matrix operand order repaired and checked by reading, not compiled, not run on hardware
//
#include <hip/hip_runtime.h>
#include <hip/hip_bf16.h>

// Problem sizes (fixed by the reference).
constexpr int B  = 4;
constexpr int K  = 16;
constexpr int CN = 256;
constexpr int H  = 64;
constexpr int W  = 64;
constexpr int HW = H * W;

typedef __attribute__((ext_vector_type(16))) __bf16 v16bf;
typedef __attribute__((ext_vector_type(8)))  __bf16 v8bf;
typedef __attribute__((ext_vector_type(8)))  float  v8f;

// ---------------------------------------------------------------------------
// Pass 1: transpose (B, CN, HW) f32  ->  (B, HW, CN) bf16 via LDS tiles.
// Coalesced 4B reads along HW, coalesced 2B writes along CN.
// ---------------------------------------------------------------------------
__global__ void transpose_to_bf16(const float* __restrict__ src,
                                  __hip_bfloat16* __restrict__ dst) {
  __shared__ float tile[16][17];  // +1 pad kills LDS bank conflicts

  int t = blockIdx.x;
  const int PT = HW / 16;         // 256 pixel tiles
  const int CT = CN / 16;         // 16 channel tiles
  int pt = t % PT; t /= PT;
  int ct = t % CT;
  int b  = t / CT;

  int tx = threadIdx.x & 15;
  int ty = threadIdx.x >> 4;

  int c = ct * 16 + ty;
  int p = pt * 16 + tx;
  tile[ty][tx] = src[((size_t)b * CN + c) * HW + p];
  __syncthreads();

  int pw = pt * 16 + ty;
  int cw = ct * 16 + tx;
  dst[((size_t)b * HW + pw) * CN + cw] = __float2bfloat16(tile[tx][ty]);
}

// ---------------------------------------------------------------------------
// Pass 2: one wave per 16-pixel tile. For each candidate k, gather 16 ref
// vectors (contiguous 512B rows in refT) as the B matrix, multiply against
// the tile's input vectors (A matrix, loaded once) with 8 chained
// v_wmma_f32_16x16x32_bf16, and read the diagonal as the correlation scores.
// Diagonal-owning lanes keep a running argmax over k, then write offsets.
// ---------------------------------------------------------------------------
__global__ void __launch_bounds__(256)
corr_argmax_wmma(const __hip_bfloat16* __restrict__ inpT,
                 const __hip_bfloat16* __restrict__ refT,
                 const float* __restrict__ aggx,
                 const float* __restrict__ aggy,
                 float* __restrict__ out) {
  const int lane = threadIdx.x & 31;
  const int wave = threadIdx.x >> 5;
  const int tile = blockIdx.x * (blockDim.x >> 5) + wave;  // 0..1023

  const int TPB = HW / 16;              // 256 tiles per batch image
  const int b   = tile / TPB;
  const int p0  = (tile % TPB) * 16;

  const int col    = lane & 15;         // matrix row (A) / column (B) index
  const int hiHalf = lane >> 4;         // lane-group split of the K dim
  const int base0  = hiHalf * 8;        // unused; A-fragment K group is also {kb0..kb0+15}
  const int kb0    = hiHalf * 16;       // B-fragment K group:  {kb0..kb0+15}

  // --- Load A fragments (input features for this pixel tile) once. ---
  const __bf16* inp_row =
      (const __bf16*)inpT + ((size_t)b * HW + p0 + col) * CN;
  v16bf afrag[8];
#pragma unroll
  for (int s = 0; s < 8; ++s) {
    const int c0 = s * 32;
    v8bf lo = *(const v8bf*)(inp_row + c0 + kb0);        // K = kb0..+7
    v8bf hi = *(const v8bf*)(inp_row + c0 + kb0 + 8);    // K = kb0+8..+15
    v16bf a;
#pragma unroll
    for (int i = 0; i < 8; ++i) { a[i] = lo[i]; a[i + 8] = hi[i]; }
    afrag[s] = a;
  }

  // Diagonal ownership: lanes 0-7 own pixels 0-7 (acc VGPR = lane),
  // lanes 24-31 own pixels 8-15 (acc VGPR = lane-24). Others own none.
  const int ownp = (lane < 8) ? lane : ((lane >= 24) ? lane - 16 : -1);
  const int ownr = (lane < 8) ? lane : (lane - 24);

  const int pcol = p0 + col;  // pixel whose coords this lane's B-column needs
  const float* axp = aggx + (size_t)b * K * HW + pcol;
  const float* ayp = aggy + (size_t)b * K * HW + pcol;

  float best  = -3.4e38f;
  int   bestk = 0;

  for (int k = 0; k < K; ++k) {
    // Gather index for this lane's column pixel at candidate k.
    const int gx = (int)axp[(size_t)k * HW];
    const int gy = (int)ayp[(size_t)k * HW];
    const int g  = gx + gy * W;

    const __bf16* ref_row =
        (const __bf16*)refT + ((size_t)b * HW + g) * CN + kb0;

    v8f acc = {};
#pragma unroll
    for (int s = 0; s < 8; ++s) {
      v16bf bfrag = *(const v16bf*)(ref_row + s * 32);  // 32B contiguous
      acc = __builtin_amdgcn_wmma_f32_16x16x32_bf16(
          /*neg_a=*/false, afrag[s], /*neg_b=*/false, bfrag,
          /*c_mod=*/(short)0, acc, /*reuse_a=*/false, /*reuse_b=*/false);
    }

    // Extract this lane's diagonal element (constant-index select chain).
    float score = acc[0];
#pragma unroll
    for (int r = 1; r < 8; ++r) score = (ownr == r) ? acc[r] : score;

    if (ownp >= 0 && score > best) { best = score; bestk = k; }  // first-max
  }

  if (ownp >= 0) {
    const int p = p0 + ownp;
    const size_t ci = ((size_t)b * K + bestk) * HW + p;
    out[(size_t)b * HW + p]                    = aggx[ci];  // offset_x
    out[(size_t)B * HW + (size_t)b * HW + p]   = aggy[ci];  // offset_y
  }
}

// ---------------------------------------------------------------------------
extern "C" void kernel_launch(void* const* d_in, const int* in_sizes, int n_in,
                              void* d_out, int out_size, void* d_ws, size_t ws_size,
                              hipStream_t stream) {
  const float* inp = (const float*)d_in[0];  // (B, CN, H, W)
  const float* ref = (const float*)d_in[1];  // (B, CN, H, W)
  const float* ax  = (const float*)d_in[2];  // (B, K, H, W)
  const float* ay  = (const float*)d_in[3];  // (B, K, H, W)
  float* out = (float*)d_out;                // offset_x ++ offset_y

  __hip_bfloat16* inpT = (__hip_bfloat16*)d_ws;                 // (B, HW, CN) bf16
  __hip_bfloat16* refT = inpT + (size_t)B * HW * CN;            // (B, HW, CN) bf16

  const int tTiles = B * (CN / 16) * (HW / 16);  // 16384 tiles of 16x16
  transpose_to_bf16<<<tTiles, 256, 0, stream>>>(inp, inpT);
  transpose_to_bf16<<<tTiles, 256, 0, stream>>>(ref, refT);

  // 1024 pixel tiles, one wave each: 128 blocks x 8 waves.
  corr_argmax_wmma<<<128, 256, 0, stream>>>(inpT, refT, ax, ay, out);
}
